// CausalSelfAttention_67619965108438
// MI455X (gfx1250) — matrix operand order repaired and checked by reading, not compiled, not run on hardware
//
#include <hip/hip_runtime.h>

typedef __attribute__((ext_vector_type(16))) _Float16 v16h;
typedef __attribute__((ext_vector_type(8)))  _Float16 v8h;
typedef __attribute__((ext_vector_type(8)))  float    v8f;

#define BB 2
#define TT 2048
#define CC 768
#define HH 12
#define HD 64
#define C3 2304
#define BT (BB*TT) /* 4096 */

// ---- WMMA fragment helpers (CDNA5 wave32 16x16x32 f16 layouts) --------------
// A-matrix (16xK, lane = M row, halves hold K):
//   VGPR0..3 : K = g*8 + 0..7 ; VGPR4..7 : K = 16 + g*8 + 0..7   (g = lane>>4)
static __device__ __forceinline__ v16h load_fragA(const _Float16* __restrict__ row,
                                                  int kbase, int g) {
  v8h lo = *(const v8h*)(row + kbase + g * 8);
  v8h hi = *(const v8h*)(row + kbase + 16 + g * 8);
  return __builtin_shufflevector(lo, hi, 0,1,2,3,4,5,6,7,8,9,10,11,12,13,14,15);
}
// B-matrix (Kx16, lane = N column, K = g*8 + 0..7, then 16 + g*8 + 0..7)
static __device__ __forceinline__ v16h load_fragB(const _Float16* __restrict__ row,
                                                  int kbase, int g) {
  v8h lo = *(const v8h*)(row + kbase + g * 8);
  v8h hi = *(const v8h*)(row + kbase + 16 + g * 8);
  return __builtin_shufflevector(lo, hi, 0,1,2,3,4,5,6,7,8,9,10,11,12,13,14,15);
}
static __device__ __forceinline__ v8f wmma_f32(v16h a, v16h b, v8f c) {
  return __builtin_amdgcn_wmma_f32_16x16x32_f16(false, a, false, b, (short)0, c,
                                                false, false);
}

// ---- elementwise prep kernels ----------------------------------------------
__global__ void k_f32_to_f16(const float* __restrict__ in,
                             _Float16* __restrict__ out, int n) {
  int i = blockIdx.x * 256 + threadIdx.x;
  if (i < n) out[i] = (_Float16)in[i];
}

// W[K][N] (f32, row-major) -> Wt[N][K] (f16) so B-fragments load contiguously.
__global__ void k_transpose_f16(const float* __restrict__ W,
                                _Float16* __restrict__ Wt, int K, int N) {
  int i = blockIdx.x * 256 + threadIdx.x;
  if (i < K * N) {
    int k = i / N, n = i % N;
    Wt[(size_t)n * K + k] = (_Float16)W[i];
  }
}

// V part of qkv [b,t,2C + h*64 + d] -> Vt[b,h,d,t] (f16)
__global__ void k_build_vt(const _Float16* __restrict__ qkv,
                           _Float16* __restrict__ Vt) {
  int i = blockIdx.x * 256 + threadIdx.x;
  if (i < BB * HH * HD * TT) {
    int t = i % TT;
    int d = (i / TT) % HD;
    int h = (i / (TT * HD)) % HH;
    int b = i / (TT * HD * HH);
    Vt[i] = qkv[((size_t)(b * TT + t)) * C3 + 2 * CC + h * HD + d];
  }
}

// ---- WMMA GEMM: out[M][N] = A[M][K] @ Bt[N][K]^T + bias ---------------------
// One wave computes a 32(M) x 64(N) block: 8 WMMAs / 12 b128 loads per k-step.
// 4 waves / 128-thread block.
__global__ __launch_bounds__(128) void k_gemm(const _Float16* __restrict__ A,
                                              const _Float16* __restrict__ Bt,
                                              const float* __restrict__ bias,
                                              _Float16* __restrict__ outH,
                                              float* __restrict__ outF,
                                              int N, int K) {
  int lane = threadIdx.x & 31, wid = threadIdx.x >> 5;
  int wave = blockIdx.x * 4 + wid;
  int nstrips = N / 64;
  int m0 = (wave / nstrips) * 32;
  int n0 = (wave % nstrips) * 64;
  int g = lane >> 4, ln = lane & 15;

  const _Float16* arow0 = A + (size_t)(m0 + ln) * K;
  const _Float16* arow1 = A + (size_t)(m0 + 16 + ln) * K;
  const _Float16* brow  = Bt + (size_t)(n0 + ln) * K;

  v8f c[2][4] = {};
  for (int kk = 0; kk < K; kk += 32) {
    // prefetch next k-tile of the streamed rows (global_prefetch_b8, no LOADcnt)
    if (kk + 64 < K) {
      __builtin_prefetch(arow0 + kk + 64, 0, 1);
      __builtin_prefetch(arow1 + kk + 64, 0, 1);
      __builtin_prefetch(brow + kk + 64, 0, 1);
    }
    v16h a0 = load_fragA(arow0, kk, g);
    v16h a1 = load_fragA(arow1, kk, g);
#pragma unroll
    for (int t = 0; t < 4; t++) {
      v16h bfrag = load_fragB(brow + (size_t)(t * 16) * K, kk, g);
      c[0][t] = wmma_f32(a0, bfrag, c[0][t]);
      c[1][t] = wmma_f32(a1, bfrag, c[1][t]);
    }
  }
#pragma unroll
  for (int i = 0; i < 2; i++) {
#pragma unroll
    for (int t = 0; t < 4; t++) {
      int ncol = n0 + t * 16 + ln;
      float bv = bias[ncol];
#pragma unroll
      for (int r = 0; r < 8; r++) {        // C-layout: m = r + g*8, n = ln
        int m = m0 + i * 16 + r + g * 8;
        float val = c[i][t][r] + bv;
        if (outH) outH[(size_t)m * N + ncol] = (_Float16)val;
        else      outF[(size_t)m * N + ncol] = val;
      }
    }
  }
}

// ---- Flash attention: one wave per (b, h, 16-row Q block) -------------------
__global__ __launch_bounds__(128) void k_attn(const _Float16* __restrict__ qkv,
                                              const _Float16* __restrict__ Vt,
                                              _Float16* __restrict__ Y) {
  __shared__ _Float16 P[4][16][40];   // padded: 80B row stride (no 4-way conflicts)
  int lane = threadIdx.x & 31, wid = threadIdx.x >> 5;
  int wave = blockIdx.x * 4 + wid;
  int qb = wave & 127;                 // T/16 = 128 blocks
  int h  = (wave >> 7) % HH;
  int b  = wave / (128 * HH);
  int q0 = qb * 16;
  int g = lane >> 4, ln = lane & 15;

  const _Float16* qrow = qkv + (size_t)(b * TT + q0 + ln) * C3 + h * HD;
  v16h qa0 = load_fragA(qrow, 0, g);
  v16h qa1 = load_fragA(qrow, 32, g);

  float mi[8], li[8];
  v8f o0 = {}, o1 = {}, o2 = {}, o3 = {};
#pragma unroll
  for (int r = 0; r < 8; r++) { mi[r] = -1e30f; li[r] = 0.f; }

  const _Float16* vbase = Vt + (size_t)((b * HH + h) * HD) * TT;

  for (int kb = 0; kb < q0 + 16; kb += 32) {      // 32 keys / iteration
    const _Float16* krow0 = qkv + (size_t)(b * TT + kb + ln) * C3 + CC + h * HD;
    const _Float16* krow1 = krow0 + (size_t)16 * C3;
    v8f s0 = {}, s1 = {};
    s0 = wmma_f32(qa0, load_fragB(krow0, 0, g), s0);
    s0 = wmma_f32(qa1, load_fragB(krow0, 32, g), s0);
    s1 = wmma_f32(qa0, load_fragB(krow1, 0, g), s1);
    s1 = wmma_f32(qa1, load_fragB(krow1, 32, g), s1);

    float p0v[8], p1v[8];
#pragma unroll
    for (int r = 0; r < 8; r++) {
      int m = q0 + r + g * 8;                       // global query row
      float x0 = (kb + ln      <= m) ? s0[r] * 0.125f : -1e30f;
      float x1 = (kb + 16 + ln <= m) ? s1[r] * 0.125f : -1e30f;
      float mx = fmaxf(x0, x1);
      mx = fmaxf(mx, __shfl_xor(mx, 1, 32));
      mx = fmaxf(mx, __shfl_xor(mx, 2, 32));
      mx = fmaxf(mx, __shfl_xor(mx, 4, 32));
      mx = fmaxf(mx, __shfl_xor(mx, 8, 32));        // row-max over 16 lanes
      float nm = fmaxf(mi[r], mx);
      float alpha = __expf(mi[r] - nm);
      float p0 = __expf(x0 - nm);
      float p1 = __expf(x1 - nm);
      float rs = p0 + p1;
      rs += __shfl_xor(rs, 1, 32);
      rs += __shfl_xor(rs, 2, 32);
      rs += __shfl_xor(rs, 4, 32);
      rs += __shfl_xor(rs, 8, 32);
      mi[r] = nm;
      li[r] = li[r] * alpha + rs;
      o0[r] *= alpha; o1[r] *= alpha; o2[r] *= alpha; o3[r] *= alpha;
      p0v[r] = p0; p1v[r] = p1;
    }

    // C-layout -> A-layout for P via per-wave LDS staging.
    __builtin_amdgcn_wave_barrier();
#pragma unroll
    for (int r = 0; r < 8; r++) {
      int m = r + g * 8;
      P[wid][m][ln]      = (_Float16)p0v[r];
      P[wid][m][16 + ln] = (_Float16)p1v[r];
    }
    __builtin_amdgcn_wave_barrier();
    asm volatile("s_wait_dscnt 0x0" ::: "memory");   // DS in-order per wave; block compiler reorder
    v16h pa;
    {
      const _Float16* prow = &P[wid][ln][0];
      v8h lo = *(const v8h*)(prow + g * 8);
      v8h hi = *(const v8h*)(prow + 16 + g * 8);
      pa = __builtin_shufflevector(lo, hi, 0,1,2,3,4,5,6,7,8,9,10,11,12,13,14,15);
    }
    __builtin_amdgcn_wave_barrier();

    o0 = wmma_f32(pa, load_fragB(vbase + (size_t)(0 * 16 + ln) * TT, kb, g), o0);
    o1 = wmma_f32(pa, load_fragB(vbase + (size_t)(1 * 16 + ln) * TT, kb, g), o1);
    o2 = wmma_f32(pa, load_fragB(vbase + (size_t)(2 * 16 + ln) * TT, kb, g), o2);
    o3 = wmma_f32(pa, load_fragB(vbase + (size_t)(3 * 16 + ln) * TT, kb, g), o3);
  }

  v8f oo[4] = {o0, o1, o2, o3};
#pragma unroll
  for (int r = 0; r < 8; r++) {
    int m = r + g * 8;
    float inv = 1.0f / li[r];
    _Float16* yrow = Y + (size_t)(b * TT + q0 + m) * CC + h * HD;
#pragma unroll
    for (int t = 0; t < 4; t++) yrow[t * 16 + ln] = (_Float16)(oo[t][r] * inv);
  }
}

// ---- host launch ------------------------------------------------------------
extern "C" void kernel_launch(void* const* d_in, const int* in_sizes, int n_in,
                              void* d_out, int out_size, void* d_ws, size_t ws_size,
                              hipStream_t stream) {
  (void)in_sizes; (void)n_in; (void)out_size; (void)ws_size;
  const float* x      = (const float*)d_in[0];
  const float* W_attn = (const float*)d_in[1];
  const float* b_attn = (const float*)d_in[2];
  const float* W_proj = (const float*)d_in[3];
  const float* b_proj = (const float*)d_in[4];
  float* out = (float*)d_out;

  char* ws = (char*)d_ws;
  size_t off = 0;
  auto alloc = [&](size_t bytes) {
    void* p = ws + off;
    off += (bytes + 255) & ~(size_t)255;
    return p;
  };
  _Float16* x_h    = (_Float16*)alloc((size_t)BT * CC * 2);            // 6.3 MB
  _Float16* WattnT = (_Float16*)alloc((size_t)C3 * CC * 2);            // 3.5 MB
  _Float16* WprojT = (_Float16*)alloc((size_t)CC * CC * 2);            // 1.2 MB
  _Float16* qkv_h  = (_Float16*)alloc((size_t)BT * C3 * 2);            // 18.9 MB
  _Float16* Vt     = (_Float16*)alloc((size_t)BB * HH * HD * TT * 2);  // 6.3 MB
  _Float16* y_h    = (_Float16*)alloc((size_t)BT * CC * 2);            // 6.3 MB

  int n;
  n = BT * CC;
  k_f32_to_f16<<<(n + 255) / 256, 256, 0, stream>>>(x, x_h, n);
  n = CC * C3;
  k_transpose_f16<<<(n + 255) / 256, 256, 0, stream>>>(W_attn, WattnT, CC, C3);
  n = CC * CC;
  k_transpose_f16<<<(n + 255) / 256, 256, 0, stream>>>(W_proj, WprojT, CC, CC);

  // qkv[BT x 3C] = x_h @ W_attn + b_attn   (f16 out)
  k_gemm<<<(BT / 32) * (C3 / 64) / 4, 128, 0, stream>>>(x_h, WattnT, b_attn,
                                                        qkv_h, nullptr, C3, CC);
  n = BB * HH * HD * TT;
  k_build_vt<<<(n + 255) / 256, 256, 0, stream>>>(qkv_h, Vt);

  // flash attention -> y_h (f16)
  k_attn<<<(BB * HH * (TT / 16)) / 4, 128, 0, stream>>>(qkv_h, Vt, y_h);

  // out[BT x C] = y_h @ W_proj + b_proj    (f32 out)
  k_gemm<<<(BT / 32) * (CC / 64) / 4, 128, 0, stream>>>(y_h, WprojT, b_proj,
                                                        nullptr, out, CC, CC);
}
